// GNNDecoder_15195594293397
// MI455X (gfx1250) — compile-verified, matrix operand order checked by reading, not run on hardware
//
#include <hip/hip_runtime.h>
#include <hip/hip_bf16.h>

// ---------------------------------------------------------------------------
// Problem constants (match reference)
// ---------------------------------------------------------------------------
#define TT    50
#define BB    64
#define NN    21
#define EE    420           // N*(N-1)
#define DD    256
#define HH    256
#define EOUT  256
#define KPAD  288           // 259 padded up to multiple of 32
#define ME    (BB*EE)       // 26880 edge rows  (divisible by 64)
#define MG    (BB*NN)       // 1344  node rows  (divisible by 64)
#define NG    768           // fused 3-gate output width

typedef __bf16 bf16;
typedef __attribute__((ext_vector_type(16))) __bf16 v16bf;
typedef __attribute__((ext_vector_type(8)))  __bf16 v8bf;
typedef __attribute__((ext_vector_type(8)))  float  v8f;

// ---------------------------------------------------------------------------
// Shared WMMA tile machinery.
//   Block tile 64x128, BK=32, 4 waves (128 thr); each wave owns 32x64
//   (2x4 fragments of 16x16) -> 8 x v_wmma_f32_16x16x32_bf16 per K-step.
//   Double-buffered LDS staging via GLOBAL_LOAD_ASYNC_TO_LDS_B128:
//   each wave issues 6 async ops/tile; async ops complete in order, so
//   s_wait_asynccnt 6 retires the current tile while the next stays in
//   flight under the WMMA burst.
// ---------------------------------------------------------------------------
#define BM        64
#define BN        128
#define BK        32
#define LDSS      40   // bf16 elements per LDS row (64B data + 16B pad)
#define ASZ       (BM * LDSS)
#define BSZ       (BN * LDSS)

__device__ __forceinline__ v16bf load_frag(const bf16* base, int lane) {
  // base -> (row 0, k 0) of a 16x32 fragment region in LDS, row stride LDSS.
  // 16-bit A-layout: lanes 0-15 hold M=lane, K 0..7 & 16..23;
  //                  lanes 16-31 hold M=lane-16, K 8..15 & 24..31.
  const int m  = lane & 15;
  const int kh = (lane >> 4) * 8;
  const bf16* p = base + m * LDSS;
  union { v16bf v; v8bf h[2]; } u;
  u.h[0] = *(const v8bf*)(p + kh);
  u.h[1] = *(const v8bf*)(p + 16 + kh);
  return u.v;
}

__device__ __forceinline__ void issue_tileA(unsigned lds, const bf16* g) {
  asm volatile(
    "global_load_async_to_lds_b128 %0, %1, off\n\t"
    "global_load_async_to_lds_b128 %0, %1, off offset:16"
    :: "v"(lds), "v"(g) : "memory");
}
__device__ __forceinline__ void issue_tileB(unsigned lds, const bf16* g) {
  asm volatile(
    "global_load_async_to_lds_b128 %0, %1, off\n\t"
    "global_load_async_to_lds_b128 %0, %1, off offset:16\n\t"
    "global_load_async_to_lds_b128 %0, %1, off offset:32\n\t"
    "global_load_async_to_lds_b128 %0, %1, off offset:48"
    :: "v"(lds), "v"(g) : "memory");
}

#define WMMA8(Ab, Bb)                                                                                   \
  do {                                                                                                  \
    v16bf a0 = load_frag((Ab) + (wm     ) * LDSS, lane);                                                \
    v16bf a1 = load_frag((Ab) + (wm + 16) * LDSS, lane);                                                \
    v16bf b0 = load_frag((Bb) + (wn     ) * LDSS, lane);                                                \
    v16bf b1 = load_frag((Bb) + (wn + 16) * LDSS, lane);                                                \
    v16bf b2 = load_frag((Bb) + (wn + 32) * LDSS, lane);                                                \
    v16bf b3 = load_frag((Bb) + (wn + 48) * LDSS, lane);                                                \
    acc[0][0] = __builtin_amdgcn_wmma_f32_16x16x32_bf16(false, a0, false, b0, (short)0, acc[0][0], false, false); \
    acc[0][1] = __builtin_amdgcn_wmma_f32_16x16x32_bf16(false, a0, false, b1, (short)0, acc[0][1], false, false); \
    acc[0][2] = __builtin_amdgcn_wmma_f32_16x16x32_bf16(false, a0, false, b2, (short)0, acc[0][2], false, false); \
    acc[0][3] = __builtin_amdgcn_wmma_f32_16x16x32_bf16(false, a0, false, b3, (short)0, acc[0][3], false, false); \
    acc[1][0] = __builtin_amdgcn_wmma_f32_16x16x32_bf16(false, a1, false, b0, (short)0, acc[1][0], false, false); \
    acc[1][1] = __builtin_amdgcn_wmma_f32_16x16x32_bf16(false, a1, false, b1, (short)0, acc[1][1], false, false); \
    acc[1][2] = __builtin_amdgcn_wmma_f32_16x16x32_bf16(false, a1, false, b2, (short)0, acc[1][2], false, false); \
    acc[1][3] = __builtin_amdgcn_wmma_f32_16x16x32_bf16(false, a1, false, b3, (short)0, acc[1][3], false, false); \
  } while (0)

// ---------------------------------------------------------------------------
// Generic WMMA GEMM:  C[M,N] = act(A[M,K] @ W[N,K]^T + bias)
// ---------------------------------------------------------------------------
template<bool RELU, bool OUT_BF16>
__global__ __launch_bounds__(128)
void gemm_wmma(const bf16* __restrict__ A, const bf16* __restrict__ W,
               const float* __restrict__ bias,
               float* __restrict__ Cf, bf16* __restrict__ Cb,
               int M, int N, int K)
{
  __shared__ __align__(16) bf16 As[2 * ASZ];
  __shared__ __align__(16) bf16 Bs[2 * BSZ];

  const int row0 = blockIdx.x * BM;
  const int col0 = blockIdx.y * BN;
  const int tid  = threadIdx.x;
  const int lane = tid & 31;
  const int wave = tid >> 5;
  const int wm   = (wave >> 1) * 32;   // wave's 32x64 sub-tile
  const int wn   = (wave & 1)  * 64;

  v8f acc[2][4];
#pragma unroll
  for (int i = 0; i < 2; ++i)
#pragma unroll
    for (int j = 0; j < 4; ++j) acc[i][j] = {};

  // staging: A tile 64x32 -> thread (row tid>>1, 16-elem half tid&1) = 32B
  //          B tile 128x32 -> thread full row tid = 64B
  const unsigned ldsA0 = (unsigned)(size_t)(As + (tid >> 1) * LDSS + (tid & 1) * 16);
  const unsigned ldsB0 = (unsigned)(size_t)(Bs + tid * LDSS);
  const unsigned ldsA1 = ldsA0 + ASZ * (unsigned)sizeof(bf16);
  const unsigned ldsB1 = ldsB0 + BSZ * (unsigned)sizeof(bf16);
  const bf16* gA = A + (size_t)(row0 + (tid >> 1)) * K + (tid & 1) * 16;
  const bf16* gB = W + (size_t)(col0 + tid) * K;

  issue_tileA(ldsA0, gA);
  issue_tileB(ldsB0, gB);

  int buf = 0;
  for (int kt = 0; kt < K; kt += BK) {
    if (kt + BK < K) {
      issue_tileA(buf ? ldsA0 : ldsA1, gA + BK);
      issue_tileB(buf ? ldsB0 : ldsB1, gB + BK);
      asm volatile("s_wait_asynccnt 0x6" ::: "memory");  // current tile landed
    } else {
      asm volatile("s_wait_asynccnt 0x0" ::: "memory");
    }
    __syncthreads();
    WMMA8(As + buf * ASZ, Bs + buf * BSZ);
    __syncthreads();    // protect buf before it is overwritten next round
    gA += BK;
    gB += BK;
    buf ^= 1;
  }

  // epilogue: C/D layout -> element (vgpr r, lane l): M = r + 8*(l>>4), N = l&15
  const int lcol = lane & 15;
  const int lhi  = lane >> 4;
#pragma unroll
  for (int i = 0; i < 2; ++i) {
#pragma unroll
    for (int j = 0; j < 4; ++j) {
      const int col = col0 + wn + j * 16 + lcol;
      const float bv = bias[col];
#pragma unroll
      for (int r = 0; r < 8; ++r) {
        const int row = row0 + wm + i * 16 + r + 8 * lhi;
        float v = acc[i][j][r] + bv;
        if (RELU) v = fmaxf(v, 0.0f);
        const size_t idx = (size_t)row * N + col;
        if (OUT_BF16) Cb[idx] = (bf16)v; else Cf[idx] = v;
      }
    }
  }
}

// ---------------------------------------------------------------------------
// Fused gather-GEMM for edge layer 1:
//   hid1[r] = relu( [h[b,send(r)] | h[b,rec(r)]] @ w1T^T + b1 ),  K = 512.
// The A tile is gathered *directly* from the node-state array by the async
// LDS loads (per-lane addresses), eliminating the materialized edge_in
// buffer entirely. Each 16-elem chunk lies wholly in the send half (K<256)
// or the rec half (K>=256) since kt is a multiple of 32.
// edge ordering: e = i*20 + jj,  sender i = e/20,  receiver j = jj + (jj>=i)
// ---------------------------------------------------------------------------
__global__ __launch_bounds__(128)
void gemm_edge1(const bf16* __restrict__ hb, const bf16* __restrict__ W,
                const float* __restrict__ bias, bf16* __restrict__ Cb)
{
  const int K = 2 * DD;      // 512
  const int N = HH;          // 256

  __shared__ __align__(16) bf16 As[2 * ASZ];
  __shared__ __align__(16) bf16 Bs[2 * BSZ];

  const int row0 = blockIdx.x * BM;
  const int col0 = blockIdx.y * BN;
  const int tid  = threadIdx.x;
  const int lane = tid & 31;
  const int wave = tid >> 5;
  const int wm   = (wave >> 1) * 32;
  const int wn   = (wave & 1)  * 64;

  v8f acc[2][4];
#pragma unroll
  for (int i = 0; i < 2; ++i)
#pragma unroll
    for (int j = 0; j < 4; ++j) acc[i][j] = {};

  const unsigned ldsA0 = (unsigned)(size_t)(As + (tid >> 1) * LDSS + (tid & 1) * 16);
  const unsigned ldsB0 = (unsigned)(size_t)(Bs + tid * LDSS);
  const unsigned ldsA1 = ldsA0 + ASZ * (unsigned)sizeof(bf16);
  const unsigned ldsB1 = ldsB0 + BSZ * (unsigned)sizeof(bf16);

  // per-thread gather source for the A tile
  const int arow   = row0 + (tid >> 1);     // edge row
  const int half16 = (tid & 1) * 16;
  const int b  = arow / EE;
  const int e  = arow % EE;
  const int si = e / (NN - 1);
  const int jj = e % (NN - 1);
  const int rj = jj + (jj >= si ? 1 : 0);
  const bf16* pS = hb + ((size_t)b * NN + si) * DD;
  const bf16* pR = hb + ((size_t)b * NN + rj) * DD;
  auto aptr = [&](int kt) {
    const int c = kt + half16;
    return (c < DD) ? (pS + c) : (pR + (c - DD));
  };
  const bf16* gB = W + (size_t)(col0 + tid) * K;

  issue_tileA(ldsA0, aptr(0));
  issue_tileB(ldsB0, gB);

  int buf = 0;
  for (int kt = 0; kt < K; kt += BK) {
    if (kt + BK < K) {
      issue_tileA(buf ? ldsA0 : ldsA1, aptr(kt + BK));
      issue_tileB(buf ? ldsB0 : ldsB1, gB + BK);
      asm volatile("s_wait_asynccnt 0x6" ::: "memory");
    } else {
      asm volatile("s_wait_asynccnt 0x0" ::: "memory");
    }
    __syncthreads();
    WMMA8(As + buf * ASZ, Bs + buf * BSZ);
    __syncthreads();
    gB += BK;
    buf ^= 1;
  }

  const int lcol = lane & 15;
  const int lhi  = lane >> 4;
#pragma unroll
  for (int i = 0; i < 2; ++i) {
#pragma unroll
    for (int j = 0; j < 4; ++j) {
      const int col = col0 + wn + j * 16 + lcol;
      const float bv = bias[col];
#pragma unroll
      for (int r = 0; r < 8; ++r) {
        const int row = row0 + wm + i * 16 + r + 8 * lhi;
        float v = fmaxf(acc[i][j][r] + bv, 0.0f);
        Cb[(size_t)row * N + col] = (bf16)v;
      }
    }
  }
}

// ---------------------------------------------------------------------------
// edge2node mean aggregate + build GRU input  ni[B*N, 288] (bf16)
//   cols 0..255 : mean of 20 incoming edge messages
//   cols 256..258: raw input(t);  259..287: zero pad
// ---------------------------------------------------------------------------
__global__ __launch_bounds__(KPAD)
void aggregate_ni(const bf16* __restrict__ e2, const float* __restrict__ inp_t,
                  bf16* __restrict__ ni)
{
  const int row = blockIdx.x;           // b*21+n
  const int b = row / NN, n = row % NN;
  const int d = threadIdx.x;            // 0..287
  if (d < EOUT) {
    float s = 0.0f;
#pragma unroll
    for (int i = 0; i < NN; ++i) {
      if (i == n) continue;
      const int e = i * (NN - 1) + (n < i ? n : n - 1);  // edge i -> n
      s += (float)e2[((size_t)b * EE + e) * EOUT + d];
    }
    ni[(size_t)row * KPAD + d] = (bf16)(s * (1.0f / (NN - 1)));
  } else if (d < EOUT + 3) {
    ni[(size_t)row * KPAD + d] = (bf16)inp_t[(size_t)row * 3 + (d - EOUT)];
  } else {
    ni[(size_t)row * KPAD + d] = (bf16)0.0f;
  }
}

// ---------------------------------------------------------------------------
// GRU elementwise update
// ---------------------------------------------------------------------------
__global__ __launch_bounds__(256)
void gru_step(const float* __restrict__ gin, const float* __restrict__ gh,
              float* __restrict__ h, bf16* __restrict__ hb,
              float* __restrict__ out_t, float* __restrict__ out_state)
{
  const int row = blockIdx.x;
  const int d   = threadIdx.x;
  const size_t g = (size_t)row * NG;
  const float r  = 1.0f / (1.0f + __expf(-(gin[g + d]           + gh[g + d])));
  const float ii = 1.0f / (1.0f + __expf(-(gin[g + DD + d]      + gh[g + DD + d])));
  const float nn = tanhf(gin[g + 2*DD + d] + r * gh[g + 2*DD + d]);
  const size_t idx = (size_t)row * DD + d;
  const float hnew = (1.0f - ii) * nn + ii * h[idx];
  h[idx]     = hnew;
  hb[idx]    = (bf16)hnew;
  out_t[idx] = hnew;
  if (out_state) out_state[idx] = hnew;
}

// ---------------------------------------------------------------------------
// prep kernels: weight transposes -> bf16 [N,K], bias concat, h init
// ---------------------------------------------------------------------------
__global__ void transpose_bf16(const float* __restrict__ w, bf16* __restrict__ wt,
                               int K, int Nw)      // w: [K,Nw] -> wt: [Nw,K]
{
  const int n = blockIdx.x;
  const int k = blockIdx.y * blockDim.x + threadIdx.x;
  if (k < K) wt[(size_t)n * K + k] = (bf16)w[(size_t)k * Nw + n];
}

__global__ __launch_bounds__(KPAD)
void build_wiT(const float* __restrict__ wr, const float* __restrict__ wi,
               const float* __restrict__ wn, bf16* __restrict__ wt)  // [768,288]
{
  const int n = blockIdx.x;             // 0..767
  const int k = threadIdx.x;            // 0..287
  const int s = n >> 8, nn = n & 255;
  const float* w = (s == 0) ? wr : (s == 1) ? wi : wn;
  wt[(size_t)n * KPAD + k] = (k < 259) ? (bf16)w[(size_t)k * DD + nn] : (bf16)0.0f;
}

__global__ __launch_bounds__(256)
void build_whT(const float* __restrict__ wr, const float* __restrict__ wi,
               const float* __restrict__ wn, bf16* __restrict__ wt)  // [768,256]
{
  const int n = blockIdx.x;
  const int k = threadIdx.x;
  const int s = n >> 8, nn = n & 255;
  const float* w = (s == 0) ? wr : (s == 1) ? wi : wn;
  wt[(size_t)n * DD + k] = (bf16)w[(size_t)k * DD + nn];
}

__global__ __launch_bounds__(256)
void init_misc(const float* __restrict__ br, const float* __restrict__ bi,
               const float* __restrict__ bn, float* __restrict__ bias_i,
               float* __restrict__ bias_z,
               const float* __restrict__ h0, float* __restrict__ h,
               bf16* __restrict__ hb, int total)
{
  const int idx = blockIdx.x * blockDim.x + threadIdx.x;
  if (idx < NG) {
    bias_i[idx] = (idx < DD) ? br[idx] : (idx < 2*DD) ? bi[idx - DD] : bn[idx - 2*DD];
    bias_z[idx] = 0.0f;
  }
  if (idx < total) { h[idx] = h0[idx]; hb[idx] = (bf16)h0[idx]; }
}

// ---------------------------------------------------------------------------
// host driver
// ---------------------------------------------------------------------------
extern "C" void kernel_launch(void* const* d_in, const int* in_sizes, int n_in,
                              void* d_out, int out_size, void* d_ws, size_t ws_size,
                              hipStream_t stream)
{
  const float* inp  = (const float*)d_in[0];   // [T,B,N,3]
  const float* h0   = (const float*)d_in[1];   // [B,N,256]
  const float* w1   = (const float*)d_in[4];   // [512,256]
  const float* b1   = (const float*)d_in[5];
  const float* w2   = (const float*)d_in[6];   // [256,256]
  const float* b2   = (const float*)d_in[7];
  const float* w_hr = (const float*)d_in[8];
  const float* w_hi = (const float*)d_in[9];
  const float* w_hn = (const float*)d_in[10];
  const float* w_ir = (const float*)d_in[11];
  const float* b_ir = (const float*)d_in[12];
  const float* w_ii = (const float*)d_in[13];
  const float* b_ii = (const float*)d_in[14];
  const float* w_in = (const float*)d_in[15];
  const float* b_in = (const float*)d_in[16];

  // workspace partition (256B aligned)
  char* p = (char*)d_ws;
  auto alloc = [&](size_t bytes) { char* r = p; p += (bytes + 255) & ~(size_t)255; return (void*)r; };
  bf16*  hid1    = (bf16*)alloc((size_t)ME * HH   * sizeof(bf16));
  bf16*  e2      = (bf16*)alloc((size_t)ME * EOUT * sizeof(bf16));
  bf16*  ni      = (bf16*)alloc((size_t)MG * KPAD * sizeof(bf16));
  float* gin     = (float*)alloc((size_t)MG * NG * sizeof(float));
  float* gh      = (float*)alloc((size_t)MG * NG * sizeof(float));
  float* hf      = (float*)alloc((size_t)MG * DD * sizeof(float));
  bf16*  hb      = (bf16*) alloc((size_t)MG * DD * sizeof(bf16));
  bf16*  w1T     = (bf16*) alloc((size_t)HH * 2*DD * sizeof(bf16));
  bf16*  w2T     = (bf16*) alloc((size_t)EOUT * HH * sizeof(bf16));
  bf16*  wiT     = (bf16*) alloc((size_t)NG * KPAD * sizeof(bf16));
  bf16*  whT     = (bf16*) alloc((size_t)NG * DD   * sizeof(bf16));
  float* bias_i  = (float*)alloc((size_t)NG * sizeof(float));
  float* bias_z  = (float*)alloc((size_t)NG * sizeof(float));

  float* out    = (float*)d_out;                        // [T,B,N,256]
  float* statep = out + (size_t)TT * MG * DD;           // final state slot

  // ---- prep (run every launch; deterministic) ----
  transpose_bf16<<<dim3(HH,   2), 256, 0, stream>>>(w1, w1T, 2*DD, HH);   // [256,512]
  transpose_bf16<<<dim3(EOUT, 1), 256, 0, stream>>>(w2, w2T, HH,   EOUT); // [256,256]
  build_wiT<<<NG, KPAD, 0, stream>>>(w_ir, w_ii, w_in, wiT);
  build_whT<<<NG, DD,   0, stream>>>(w_hr, w_hi, w_hn, whT);
  {
    const int total = MG * DD;
    init_misc<<<(total + 255) / 256, 256, 0, stream>>>(b_ir, b_ii, b_in, bias_i,
                                                       bias_z, h0, hf, hb, total);
  }

  // ---- scan over T steps ----
  for (int t = 0; t < TT; ++t) {
    // fused node2edge gather + edge MLP layer 1 (async gather-GEMM)
    gemm_edge1<<<dim3(ME/BM, HH/BN), 128, 0, stream>>>(hb, w1T, b1, hid1);

    gemm_wmma<true,  true ><<<dim3(ME/BM, EOUT/BN), 128, 0, stream>>>(
        hid1, w2T, b2, nullptr, e2, ME, EOUT, HH);

    aggregate_ni<<<MG, KPAD, 0, stream>>>(e2, inp + (size_t)t * MG * 3, ni);

    gemm_wmma<false, false><<<dim3(MG/BM, NG/BN), 128, 0, stream>>>(
        ni, wiT, bias_i, gin, nullptr, MG, NG, KPAD);
    gemm_wmma<false, false><<<dim3(MG/BM, NG/BN), 128, 0, stream>>>(
        hb, whT, bias_z, gh, nullptr, MG, NG, DD);

    gru_step<<<MG, DD, 0, stream>>>(gin, gh, hf, hb,
                                    out + (size_t)t * MG * DD,
                                    (t == TT - 1) ? statep : nullptr);
  }
}
